// MemEffAttention_77498389889299
// MI455X (gfx1250) — matrix operand order repaired and checked by reading, not compiled, not run on hardware
//
#include <hip/hip_runtime.h>
#include <hip/hip_bf16.h>
#include <math.h>

typedef __attribute__((ext_vector_type(16))) _Float16 v16h;
typedef __attribute__((ext_vector_type(8)))  _Float16 v8h;
typedef __attribute__((ext_vector_type(8)))  float    v8f;
typedef __attribute__((ext_vector_type(4)))  float    v4f;
typedef __attribute__((ext_vector_type(4)))  unsigned int v4u;
typedef __attribute__((ext_vector_type(8)))  int      v8i;
typedef __attribute__((ext_vector_type(4)))  int      v4i;

#define B_  4
#define N_  2048
#define C_  1024
#define H_  16
#define D_  64
#define SCALE_ 0.125f   // 64^-0.5
#define LDST  40        // f16 LDS row stride in halves (32 data + 4-dword pad)
#define LDSTF 36        // f32 LDS row stride in floats (32 data + 4-dword pad)

#if __has_builtin(__builtin_amdgcn_tensor_load_to_lds) && \
    __has_builtin(__builtin_amdgcn_s_wait_tensorcnt)
#define USE_TDM 1
#else
#define USE_TDM 0
#endif

// ---------------------------------------------------------------------------
// Fragment loaders per CDNA5 ISA 7.12.2 (wave32, 16-bit A/B layouts).
// A 16x32: lane L holds row M=L&15; halves K=[s,s+8) and [s+16,s+24), s=(L&16)?8:0
// B 32x16: lane L holds col N=L&15; halves K=[t,t+8) and [t+16,t+24), t=(L&16)?8:0
// ---------------------------------------------------------------------------
__device__ __forceinline__ v16h frag_a(const _Float16* row, int lane) {
    const int s = (lane & 16) ? 8 : 0;
    v8h lo = *(const v8h*)(row + s);
    v8h hi = *(const v8h*)(row + s + 16);
    return __builtin_shufflevector(lo, hi, 0,1,2,3,4,5,6,7,8,9,10,11,12,13,14,15);
}
__device__ __forceinline__ v16h frag_b(const _Float16* col, int lane) {
    const int t = (lane & 16) ? 8 : 0;
    v8h lo = *(const v8h*)(col + t);
    v8h hi = *(const v8h*)(col + t + 16);
    return __builtin_shufflevector(lo, hi, 0,1,2,3,4,5,6,7,8,9,10,11,12,13,14,15);
}
// Same fragments built from an fp32 LDS tile (convert at use; TDM is raw copy).
__device__ __forceinline__ v16h frag_a_f32(const float* row, int lane) {
    const int s = (lane & 16) ? 8 : 0;
    v4f a0 = *(const v4f*)(row + s);
    v4f a1 = *(const v4f*)(row + s + 4);
    v4f b0 = *(const v4f*)(row + s + 16);
    v4f b1 = *(const v4f*)(row + s + 20);
    v16h r;
#pragma unroll
    for (int i = 0; i < 4; ++i) {
        r[i]      = (_Float16)a0[i];
        r[i + 4]  = (_Float16)a1[i];
        r[i + 8]  = (_Float16)b0[i];
        r[i + 12] = (_Float16)b1[i];
    }
    return r;
}
__device__ __forceinline__ v16h frag_b_f32(const float* col, int lane) {
    const int t = (lane & 16) ? 8 : 0;
    v4f a0 = *(const v4f*)(col + t);
    v4f a1 = *(const v4f*)(col + t + 4);
    v4f a2 = *(const v4f*)(col + t + 16);
    v4f a3 = *(const v4f*)(col + t + 20);
    v16h r;
#pragma unroll
    for (int i = 0; i < 4; ++i) {
        r[i]      = (_Float16)a0[i];
        r[i + 4]  = (_Float16)a1[i];
        r[i + 8]  = (_Float16)a2[i];
        r[i + 12] = (_Float16)a3[i];
    }
    return r;
}
__device__ __forceinline__ v8f wmma_f16(v16h a, v16h b, v8f c) {
    return __builtin_amdgcn_wmma_f32_16x16x32_f16(false, a, false, b, (short)0, c,
                                                  false, false);
}

#if USE_TDM
// ---------------------------------------------------------------------------
// TDM 2D tile load: tile_d0 x tile_d1 elements from a row-major tensor
// (stride_elems per row) into LDS at lds_off, with hardware row padding.
// Descriptor packing per CDNA5 ISA 8.3/8.4. data_size_log: 1=f16, 2=f32.
// pad_interval: pad after (8B << pi); pad_amount: (pa+1) dwords.
// ---------------------------------------------------------------------------
__device__ __forceinline__ void
tdm_load_2d(unsigned lds_off, const void* gptr, unsigned tile_d0,
            unsigned tile_d1, unsigned stride_elems, unsigned data_size_log,
            unsigned pad_interval, unsigned pad_amount) {
    const unsigned long long ga = (unsigned long long)gptr;
    v4u g0;
    g0[0] = 1u;                                        // count=1, user desc
    g0[1] = lds_off;                                   // lds_addr [63:32]
    g0[2] = (unsigned)(ga & 0xffffffffu);              // global_addr lo
    g0[3] = (unsigned)((ga >> 32) & 0x01ffffffu)       // global_addr hi (57b)
            | (2u << 30);                              // type=2 (image)
    const unsigned td0 = 1u << 30, td1 = 1u << 30;     // huge: tiles never OOB
    v8i g1;
    g1[0] = (int)((data_size_log << 16) | (1u << 20)   // data_size | pad_enable
                  | (pad_interval << 22) | (pad_amount << 25));
    g1[1] = (int)((td0 & 0xffffu) << 16);              // tensor_dim0 lo16
    g1[2] = (int)((td0 >> 16) | ((td1 & 0xffffu) << 16));
    g1[3] = (int)((td1 >> 16) | (tile_d0 << 16));      // tile_dim0
    g1[4] = (int)(tile_d1 & 0xffffu);                  // tile_dim1 (tile_dim2=0)
    g1[5] = (int)stride_elems;                         // tensor_dim0_stride lo32
    g1[6] = 0;
    g1[7] = 0;
    const v4i z4 = {0, 0, 0, 0};
#if __clang_major__ >= 23
    const v8i z8 = {0, 0, 0, 0, 0, 0, 0, 0};
    __builtin_amdgcn_tensor_load_to_lds(g0, g1, z4, z4, z8, 0);
#else
    __builtin_amdgcn_tensor_load_to_lds(g0, g1, z4, z4, 0);
#endif
}
#define LDS_OFF(p) ((unsigned)(unsigned long long)(const void*)(p))
#endif // USE_TDM

// Manual-fill fallbacks (also used when TDM builtins are unavailable).
__device__ __forceinline__ void fill_tile_f32(float* lds, const float* g,
                                              int gstride, int tid) {
    const int r = tid >> 1, c = (tid & 1) * 16;
    const float* src = g + (size_t)r * gstride + c;
    v4f f0 = *(const v4f*)(src + 0);
    v4f f1 = *(const v4f*)(src + 4);
    v4f f2 = *(const v4f*)(src + 8);
    v4f f3 = *(const v4f*)(src + 12);
    *(v4f*)(lds + r * LDSTF + c + 0)  = f0;
    *(v4f*)(lds + r * LDSTF + c + 4)  = f1;
    *(v4f*)(lds + r * LDSTF + c + 8)  = f2;
    *(v4f*)(lds + r * LDSTF + c + 12) = f3;
}
__device__ __forceinline__ void fill_tile_f16(_Float16* lds, const _Float16* g,
                                              int gstride, int tid) {
    const int r = tid >> 1, c = (tid & 1) * 16;
    const _Float16* src = g + (size_t)r * gstride + c;
    *(v8h*)(lds + r * LDST + c)     = *(const v8h*)(src + 0);
    *(v8h*)(lds + r * LDST + c + 8) = *(const v8h*)(src + 8);
}

// ---------------------------------------------------------------------------
// Kernel 1: qkv = x @ qkv_w.T (8192 x 3072), fused +q_g/+k_g bias, *scale on q,
// scattered to f16 workspace:  q,k -> (B,H,N,D);  v -> (B,H,D,N) transposed.
// grid = (48, 128), block = 128 (4 waves). A/B tiles double-buffered fp32 in
// LDS via TDM (pad_interval=4 -> 128B rows + 4-dword pad = 36-float stride).
// ---------------------------------------------------------------------------
__global__ void __launch_bounds__(128)
qkv_wmma_kernel(const float* __restrict__ x, const float* __restrict__ w,
                const float* __restrict__ g_info,
                _Float16* __restrict__ qf, _Float16* __restrict__ kf,
                _Float16* __restrict__ vt) {
    __shared__ float At[2][64 * LDSTF];
    __shared__ float Bt[2][64 * LDSTF];
    const int tid  = threadIdx.x;
    const int wave = tid >> 5, lane = tid & 31;
    const int rowbase = blockIdx.y * 64;       // flattened (b, token)
    const int colbase = blockIdx.x * 64;       // qkv channel
    const float* Abase = x + (size_t)rowbase * C_;
    const float* Bbase = w + (size_t)colbase * C_;

    v8f acc[4] = {};
    int cur = 0;
#if USE_TDM
    if (wave == 0) {
        tdm_load_2d(LDS_OFF(At[0]), Abase, 32, 64, C_, 2, 4, 3);
        tdm_load_2d(LDS_OFF(Bt[0]), Bbase, 32, 64, C_, 2, 4, 3);
    }
#endif
    for (int k0 = 0; k0 < C_; k0 += 32) {
#if USE_TDM
        if (wave == 0) {
            if (k0 + 32 < C_) {
                tdm_load_2d(LDS_OFF(At[cur ^ 1]), Abase + k0 + 32, 32, 64, C_, 2, 4, 3);
                tdm_load_2d(LDS_OFF(Bt[cur ^ 1]), Bbase + k0 + 32, 32, 64, C_, 2, 4, 3);
                __builtin_amdgcn_s_wait_tensorcnt(2);
            } else {
                __builtin_amdgcn_s_wait_tensorcnt(0);
            }
        }
        __syncthreads();
#else
        __syncthreads();
        fill_tile_f32(At[cur], Abase + k0, C_, tid);
        fill_tile_f32(Bt[cur], Bbase + k0, C_, tid);
        __syncthreads();
#endif
        const v16h a =
            frag_a_f32(At[cur] + (wave * 16 + (lane & 15)) * LDSTF, lane);
#pragma unroll
        for (int j = 0; j < 4; ++j) {
            const v16h b =
                frag_b_f32(Bt[cur] + (j * 16 + (lane & 15)) * LDSTF, lane);
            acc[j] = wmma_f16(a, b, acc[j]);
        }
        __syncthreads();
        cur ^= 1;
    }

    const int bb = rowbase >> 11;              // batch
    const int rhalf = (lane & 16) ? 8 : 0;
#pragma unroll
    for (int j = 0; j < 4; ++j) {
        const int cg  = colbase + j * 16 + (lane & 15);
        const int seg = cg >> 10;              // 0=q 1=k 2=v
        const int c   = cg & (C_ - 1);
        const int h   = c >> 6, d = c & (D_ - 1);
        const float bias = (seg == 0) ? g_info[c]
                         : (seg == 1) ? g_info[C_ + c] : 0.0f;
#pragma unroll
        for (int i = 0; i < 8; ++i) {
            const int tok = (rowbase & (N_ - 1)) + wave * 16 + i + rhalf;
            const float v = acc[j][i] + bias;
            const size_t bh = (size_t)(bb * H_ + h);
            if (seg == 0)
                qf[(bh * N_ + tok) * D_ + d] = (_Float16)(v * SCALE_);
            else if (seg == 1)
                kf[(bh * N_ + tok) * D_ + d] = (_Float16)v;
            else
                vt[(bh * D_ + d) * N_ + tok] = (_Float16)v;
        }
    }
}

// ---------------------------------------------------------------------------
// Kernel 2: flash attention per (b*h, 64-query tile). block = 128 (4 waves),
// wave owns 16 query rows; keys streamed 32 at a time; online softmax with
// shfl_xor reductions over the 16-lane C-matrix halves; P staged via LDS.
// ---------------------------------------------------------------------------
__global__ void __launch_bounds__(128)
attn_wmma_kernel(const _Float16* __restrict__ qf, const _Float16* __restrict__ kf,
                 const _Float16* __restrict__ vt, _Float16* __restrict__ ao) {
    __shared__ _Float16 Plds[4 * 16 * LDST];
    const int wave = threadIdx.x >> 5, lane = threadIdx.x & 31;
    const int bh = blockIdx.x;
    const int bb = bh >> 4, h = bh & (H_ - 1);
    const _Float16* Q  = qf + (size_t)bh * N_ * D_;
    const _Float16* K  = kf + (size_t)bh * N_ * D_;
    const _Float16* Vt = vt + (size_t)bh * D_ * N_;

    const int qrow = blockIdx.y * 64 + wave * 16 + (lane & 15);
    const v16h q0 = frag_a(Q + (size_t)qrow * D_ + 0,  lane);
    const v16h q1 = frag_a(Q + (size_t)qrow * D_ + 32, lane);

    v8f acc[4] = {};
    float mrow[8], lrow[8];
#pragma unroll
    for (int i = 0; i < 8; ++i) { mrow[i] = -3.0e30f; lrow[i] = 0.0f; }

    _Float16* P = Plds + wave * 16 * LDST;
    const int rhalf = (lane & 16) ? 8 : 0;

    for (int kb = 0; kb < N_; kb += 32) {
        v8f s0 = {}, s1 = {};
        {
            const size_t keyA = (size_t)(kb + (lane & 15)) * D_;
            const size_t keyB = (size_t)(kb + 16 + (lane & 15)) * D_;
            s0 = wmma_f16(q0, frag_b(K + keyA + 0,  lane), s0);
            s0 = wmma_f16(q1, frag_b(K + keyA + 32, lane), s0);
            s1 = wmma_f16(q0, frag_b(K + keyB + 0,  lane), s1);
            s1 = wmma_f16(q1, frag_b(K + keyB + 32, lane), s1);
        }
        float corr[8];
#pragma unroll
        for (int i = 0; i < 8; ++i) {
            float tm = fmaxf(s0[i], s1[i]);
            tm = fmaxf(tm, __shfl_xor(tm, 1));
            tm = fmaxf(tm, __shfl_xor(tm, 2));
            tm = fmaxf(tm, __shfl_xor(tm, 4));
            tm = fmaxf(tm, __shfl_xor(tm, 8));
            const float mn = fmaxf(mrow[i], tm);
            corr[i] = __expf(mrow[i] - mn);
            mrow[i] = mn;
            const float p0 = __expf(s0[i] - mn);
            const float p1 = __expf(s1[i] - mn);
            s0[i] = p0; s1[i] = p1;
            float rs = p0 + p1;
            rs += __shfl_xor(rs, 1);
            rs += __shfl_xor(rs, 2);
            rs += __shfl_xor(rs, 4);
            rs += __shfl_xor(rs, 8);
            lrow[i] = lrow[i] * corr[i] + rs;
        }
#pragma unroll
        for (int t = 0; t < 4; ++t)
#pragma unroll
            for (int i = 0; i < 8; ++i) acc[t][i] *= corr[i];

        {   // C-layout -> A-fragment layout via wave-private LDS tile
            const int n = lane & 15;
#pragma unroll
            for (int i = 0; i < 8; ++i) {
                P[(rhalf + i) * LDST + n]      = (_Float16)s0[i];
                P[(rhalf + i) * LDST + 16 + n] = (_Float16)s1[i];
            }
        }
        asm volatile("s_wait_dscnt 0" ::: "memory");
        const v16h pf = frag_a(P + (lane & 15) * LDST, lane);
#pragma unroll
        for (int t = 0; t < 4; ++t) {
            const v16h vf =
                frag_b(Vt + (size_t)(t * 16 + (lane & 15)) * N_ + kb, lane);
            acc[t] = wmma_f16(pf, vf, acc[t]);
        }
    }

    float inv[8];
#pragma unroll
    for (int i = 0; i < 8; ++i) inv[i] = 1.0f / lrow[i];
#pragma unroll
    for (int t = 0; t < 4; ++t) {
        const int c = h * D_ + t * 16 + (lane & 15);
#pragma unroll
        for (int i = 0; i < 8; ++i) {
            const int tok = blockIdx.y * 64 + wave * 16 + i + rhalf;
            ao[((size_t)bb * N_ + tok) * C_ + c] = (_Float16)(acc[t][i] * inv[i]);
        }
    }
}

// ---------------------------------------------------------------------------
// Kernel 3: out = attn(8192x1024 f16) @ proj_w.T + proj_b -> fp32 d_out.
// A tile DMA'd as f16 (pad_interval=3 -> 40-half rows); B tile as f32.
// ---------------------------------------------------------------------------
__global__ void __launch_bounds__(128)
proj_wmma_kernel(const _Float16* __restrict__ ao, const float* __restrict__ w,
                 const float* __restrict__ bias, float* __restrict__ out) {
    __shared__ _Float16 At[2][64 * LDST];
    __shared__ float    Bt[2][64 * LDSTF];
    const int tid  = threadIdx.x;
    const int wave = tid >> 5, lane = tid & 31;
    const int rowbase = blockIdx.y * 64;
    const int colbase = blockIdx.x * 64;
    const _Float16* Abase = ao + (size_t)rowbase * C_;
    const float*    Bbase = w + (size_t)colbase * C_;

    v8f acc[4] = {};
    int cur = 0;
#if USE_TDM
    if (wave == 0) {
        tdm_load_2d(LDS_OFF(At[0]), Abase, 32, 64, C_, 1, 3, 3);
        tdm_load_2d(LDS_OFF(Bt[0]), Bbase, 32, 64, C_, 2, 4, 3);
    }
#endif
    for (int k0 = 0; k0 < C_; k0 += 32) {
#if USE_TDM
        if (wave == 0) {
            if (k0 + 32 < C_) {
                tdm_load_2d(LDS_OFF(At[cur ^ 1]), Abase + k0 + 32, 32, 64, C_, 1, 3, 3);
                tdm_load_2d(LDS_OFF(Bt[cur ^ 1]), Bbase + k0 + 32, 32, 64, C_, 2, 4, 3);
                __builtin_amdgcn_s_wait_tensorcnt(2);
            } else {
                __builtin_amdgcn_s_wait_tensorcnt(0);
            }
        }
        __syncthreads();
#else
        __syncthreads();
        fill_tile_f16(At[cur], Abase + k0, C_, tid);
        fill_tile_f32(Bt[cur], Bbase + k0, C_, tid);
        __syncthreads();
#endif
        const v16h a = frag_a(At[cur] + (wave * 16 + (lane & 15)) * LDST, lane);
#pragma unroll
        for (int j = 0; j < 4; ++j) {
            const v16h b =
                frag_b_f32(Bt[cur] + (j * 16 + (lane & 15)) * LDSTF, lane);
            acc[j] = wmma_f16(a, b, acc[j]);
        }
        __syncthreads();
        cur ^= 1;
    }

    const int rhalf = (lane & 16) ? 8 : 0;
#pragma unroll
    for (int j = 0; j < 4; ++j) {
        const int col = colbase + j * 16 + (lane & 15);
        const float pb = bias[col];
#pragma unroll
        for (int i = 0; i < 8; ++i) {
            const int row = rowbase + wave * 16 + i + rhalf;
            out[(size_t)row * C_ + col] = acc[j][i] + pb;
        }
    }
}

// Kernel 4: pass g_info[1] (4096 f32) through to the output tail.
__global__ void tail_copy_kernel(const float* __restrict__ g_info,
                                 float* __restrict__ out_tail) {
    const int i = blockIdx.x * blockDim.x + threadIdx.x;
    if (i < 4 * C_) out_tail[i] = g_info[4 * C_ + i];
}

// ---------------------------------------------------------------------------
extern "C" void kernel_launch(void* const* d_in, const int* in_sizes, int n_in,
                              void* d_out, int out_size, void* d_ws, size_t ws_size,
                              hipStream_t stream) {
    (void)in_sizes; (void)n_in; (void)out_size; (void)ws_size;
    const float* x      = (const float*)d_in[0];   // (4,2048,1024)
    const float* g_info = (const float*)d_in[1];   // (2,4,1024)
    const float* qkv_w  = (const float*)d_in[2];   // (3072,1024)
    const float* proj_w = (const float*)d_in[3];   // (1024,1024)
    const float* proj_b = (const float*)d_in[4];   // (1024,)
    float* out = (float*)d_out;

    const size_t TEN = (size_t)B_ * H_ * N_ * D_;  // 8,388,608 halves (16 MB)
    _Float16* ws = (_Float16*)d_ws;
    _Float16* qf = ws;                             // (B,H,N,D) f16
    _Float16* kf = ws + TEN;                       // (B,H,N,D) f16
    _Float16* vt = ws + 2 * TEN;                   // (B,H,D,N) f16 (transposed)
    _Float16* ao = ws + 3 * TEN;                   // (B,N,C)   f16

    qkv_wmma_kernel<<<dim3(3 * C_ / 64, B_ * N_ / 64), 128, 0, stream>>>(
        x, qkv_w, g_info, qf, kf, vt);
    attn_wmma_kernel<<<dim3(B_ * H_, N_ / 64), 128, 0, stream>>>(qf, kf, vt, ao);
    proj_wmma_kernel<<<dim3(C_ / 64, B_ * N_ / 64), 128, 0, stream>>>(
        ao, proj_w, proj_b, out);
    tail_copy_kernel<<<dim3(16), 256, 0, stream>>>(g_info,
                                                   out + (size_t)B_ * N_ * C_);
}
